// Model_14448269984014
// MI455X (gfx1250) — compile-verified, matrix operand order checked by reading, not run on hardware
//
#include <hip/hip_runtime.h>
#include <cstdint>

#define DEVINL __device__ __forceinline__

typedef __attribute__((ext_vector_type(16))) __bf16 v16bf;
typedef __attribute__((ext_vector_type(8)))  float  v8f;

union FragAB { v16bf v; uint32_t u[8]; uint4 q[2]; };
union FragC  { v8f v; float f[8]; };

// Pack two f32 into packed bf16 (round-to-nearest-even).
DEVINL uint32_t pack2bf(float a, float b) {
  uint32_t ua = __float_as_uint(a), ub = __float_as_uint(b);
  ua += 0x7FFFu + ((ua >> 16) & 1u);
  ub += 0x7FFFu + ((ub >> 16) & 1u);
  return (ua >> 16) | (ub & 0xFFFF0000u);
}

DEVINL v8f wmma_bf16(v16bf a, v16bf b, v8f c) {
  // D = A(16x32 bf16) * B(32x16 bf16) + C(16x16 f32)
  return __builtin_amdgcn_wmma_f32_16x16x32_bf16(false, a, false, b, (short)0, c,
                                                 false, false);
}

// A fragment (16x32 bf16) from row-major bf16 tile in LDS.
// ISA layout: lane L -> m = L%16, g = L/16; VGPR0..3 = K[8g..8g+7], VGPR4..7 = K[16+8g..16+8g+7].
DEVINL void load_a_frag(FragAB& f, const char* base, int row0, int row_bytes,
                        int col_bytes, int lane) {
  const int m = lane & 15, g = lane >> 4;
  const char* p = base + (size_t)(row0 + m) * row_bytes + col_bytes + 16 * g;
  f.q[0] = *(const uint4*)(p);
  f.q[1] = *(const uint4*)(p + 32);
}

// B tiles are stored in LDS in *fragment-native* order: per 16x32 bf16 tile a
// 1KB region where lane L's fragment words u[0..3] sit at byte L*16 and
// u[4..7] at byte 512 + L*16  ->  B-frag load = two ds_load_b128, no shuffles.
DEVINL void load_b_frag_n(FragAB& f, const char* tile_base, int lane) {
  const char* p = tile_base + lane * 16;
  f.q[0] = *(const uint4*)(p);
  f.q[1] = *(const uint4*)(p + 512);
}

// Inverse map for staging: packed word (k2 in [0,16), nloc in [0,16)) of a
// tile -> word index inside the 1KB tile region.
// u[v] of lane L holds k2 = (v&3) + 4*(L>>4) + 8*(v>=4), n = L&15.
DEVINL int bfrag_word(int k2, int nloc) {
  const int low2 = k2 & 3, g = (k2 >> 2) & 1, hi = (k2 >> 3) & 1;
  return hi * 128 + (nloc + 16 * g) * 4 + low2;
}

// ---------------------------------------------------------------------------
// Kernel 1: p = sigmoid(tanh(h_t @ w1 + b1) @ w2 + b2); start = round(p - W)
// Rows ordered as decoder memory: i = t*B + b. M=64 rows per block.
// ---------------------------------------------------------------------------
__global__ __launch_bounds__(256) void mono_mlp_kernel(
    const float* __restrict__ dec, const float* __restrict__ w1,
    const float* __restrict__ b1v, const float* __restrict__ w2v,
    const float* __restrict__ b2v, const int* __restrict__ lengths,
    float* __restrict__ p_ws, int* __restrict__ start_ws)
{
  constexpr int H = 1024, H2 = 512, MB = 64, Bsz = 32, Wc = 64;
  __shared__ __align__(16) char smem[MB * 80 + 16 * H2 * 4]; // A tile + w1 chunk frags
  __shared__ float red[MB];
  char* A_sh = smem;                            // [64 rows][40 bf16] (80B rows)
  uint32_t* Bp = (uint32_t*)(smem + MB * 80);   // 32 tiles x 1KB, fragment-native

  const int tid = threadIdx.x, lane = tid & 31, w = tid >> 5;
  const int g = lane >> 4, l16 = lane & 15;
  const int i0 = blockIdx.x * MB;
  if (tid < MB) red[tid] = 0.f;

  const int mt  = w & 3;            // M-tile 0..3
  const int nt0 = (w >> 2) * 16;    // 16 N-tiles per wave

  FragC acc[16];
#pragma unroll
  for (int j = 0; j < 16; ++j)
#pragma unroll
    for (int v = 0; v < 8; ++v) acc[j].f[v] = 0.f;

  for (int kc = 0; kc < H; kc += 32) {
    __syncthreads();
    // Stage A: 64 rows x 32 k of decoder rows (t*B+b memory order), f32 -> bf16.
    for (int idx = tid; idx < MB * 16; idx += 256) {
      int r = idx >> 4, c2 = idx & 15;
      float2 f2 = *(const float2*)(dec + (size_t)(i0 + r) * H + kc + 2 * c2);
      ((uint32_t*)A_sh)[r * 20 + c2] = pack2bf(f2.x, f2.y);
    }
    // Stage B: w1 chunk [32 k][512 n] -> fragment-native bf16 tiles.
    for (int idx = tid; idx < 16 * H2; idx += 256) {
      int k2 = idx >> 9, n = idx & (H2 - 1);
      float f0 = w1[(size_t)(kc + 2 * k2) * H2 + n];
      float f1 = w1[(size_t)(kc + 2 * k2 + 1) * H2 + n];
      Bp[(n >> 4) * 256 + bfrag_word(k2, n & 15)] = pack2bf(f0, f1);
    }
    __syncthreads();
    FragAB af; load_a_frag(af, A_sh, mt * 16, 80, 0, lane);
    // Ping-pong double buffer, loads land directly in fragment registers.
    FragAB bb[2];
    load_b_frag_n(bb[0], (const char*)Bp + nt0 * 1024, lane);
#pragma unroll
    for (int j = 0; j < 16; ++j) {
      if (j + 1 < 16)
        load_b_frag_n(bb[(j + 1) & 1], (const char*)Bp + (nt0 + j + 1) * 1024, lane);
      acc[j].v = wmma_bf16(af.v, bb[j & 1].v, acc[j].v);
    }
  }

  // Epilogue: out_row[m] = sum_n tanh(z[m][n] + b1[n]) * w2[n]
  float ls[8];
#pragma unroll
  for (int v = 0; v < 8; ++v) ls[v] = 0.f;
#pragma unroll
  for (int j = 0; j < 16; ++j) {
    int n = (nt0 + j) * 16 + l16;
    float bb = b1v[n], ww = w2v[n];
#pragma unroll
    for (int v = 0; v < 8; ++v)
      ls[v] += tanhf(acc[j].f[v] + bb) * ww;
  }
#pragma unroll
  for (int v = 0; v < 8; ++v)
    atomicAdd(&red[mt * 16 + v + 8 * g], ls[v]);  // ds_add_f32
  __syncthreads();
  if (tid < MB) {
    int i = i0 + tid;
    int b = i & (Bsz - 1);
    float z   = red[tid] + b2v[0];
    float sig = 1.f / (1.f + expf(-z));
    float pv  = (float)Wc + (float)lengths[b] * sig;
    p_ws[i]     = pv;
    start_ws[i] = (int)rintf(pv - (float)Wc);
  }
}

// ---------------------------------------------------------------------------
// Kernel 2: fused score GEMM + masked softmax*gaussian + context GEMM.
// Grid (T/64, B), 288 threads = 9 waves. selection rows = enc[start_last+k].
// ---------------------------------------------------------------------------
__global__ __launch_bounds__(288) void mono_attn_kernel(
    const float* __restrict__ enc, const float* __restrict__ dec,
    const float* __restrict__ p_ws, const int* __restrict__ start_ws,
    const int* __restrict__ lengths, float* __restrict__ out)
{
  constexpr int H = 1024, Bsz = 32, T = 512, Wc = 64;
  constexpr int NP = 144;   // 129 padded to 9 tiles for score
  constexpr int KP = 160;   // 129 padded to 5 K32-chunks for context
  __shared__ __align__(16) char smem[58624];
  char*     A_sh  = smem;                         // phase A: [64][40 bf16]  (5120 B)
  char*     BpA   = smem + 5120;                  // phase A: 9 tiles x 1KB  (9216 B)
  float*    score = (float*)smem;                 // epilogue:[64][145] f32  (37120 B)
  char*     a_sh  = smem + 37120;                 // [64][168 bf16]          (21504 B)
  char*     BpB   = smem;                         // phase B: 20 tiles x 1KB (20480 B)

  const int tid = threadIdx.x, lane = tid & 31, w = tid >> 5; // w in 0..8
  const int g = lane >> 4, l16 = lane & 15;
  const int t0 = blockIdx.x * 64;
  const int b  = blockIdx.y;
  const int s0 = start_ws[(T - 1) * Bsz + b];     // selection base (start at last t)

  // -------- Phase A: score[64][129] = h_t_tile (64xH) . selection^T --------
  FragC acc[4];
#pragma unroll
  for (int mt = 0; mt < 4; ++mt)
#pragma unroll
    for (int v = 0; v < 8; ++v) acc[mt].f[v] = 0.f;

  for (int kc = 0; kc < H; kc += 32) {
    __syncthreads();
    for (int idx = tid; idx < 64 * 16; idx += 288) {   // A tile
      int r = idx >> 4, c2 = idx & 15;
      float2 f2 = *(const float2*)(dec + ((size_t)(t0 + r) * Bsz + b) * H + kc + 2 * c2);
      ((uint32_t*)A_sh)[r * 20 + c2] = pack2bf(f2.x, f2.y);
    }
    for (int idx = tid; idx < 16 * NP; idx += 288) {   // selection chunk (B matrix)
      int n = idx >> 4, k2 = idx & 15;                 // n = window index w
      uint32_t pk = 0;
      if (n <= 128) {
        float2 f2 = *(const float2*)(enc + ((size_t)(s0 + n) * Bsz + b) * H + kc + 2 * k2);
        pk = pack2bf(f2.x, f2.y);
      }
      ((uint32_t*)BpA)[(n >> 4) * 256 + bfrag_word(k2, n & 15)] = pk;
    }
    __syncthreads();
    FragAB bf; load_b_frag_n(bf, BpA + w * 1024, lane);   // wave w owns N-tile w
    FragAB aa[2];
    load_a_frag(aa[0], A_sh, 0, 80, 0, lane);
#pragma unroll
    for (int mtl = 0; mtl < 4; ++mtl) {
      if (mtl + 1 < 4) load_a_frag(aa[(mtl + 1) & 1], A_sh, (mtl + 1) * 16, 80, 0, lane);
      acc[mtl].v = wmma_bf16(aa[mtl & 1].v, bf.v, acc[mtl].v);
    }
  }
  __syncthreads();
#pragma unroll
  for (int mt = 0; mt < 4; ++mt)
#pragma unroll
    for (int v = 0; v < 8; ++v)
      score[(size_t)(mt * 16 + v + 8 * g) * 145 + w * 16 + l16] = acc[mt].f[v];
  __syncthreads();

  // -------- Epilogue: masked softmax (EPS inside softmax) * gaussian -------
  if (tid < 64) {
    const int r = tid, t = t0 + r, i = t * Bsz + b;
    const float pv = p_ws[i];
    const int s = start_ws[i], li = lengths[b];
    const int lo = Wc - s, hi = li + Wc - s;   // mask: k < lo || k >= hi
    float* row = score + (size_t)r * 145;
    float mx = -3.4e38f;
#pragma unroll 4
    for (int k = 0; k < 129; ++k) {
      float val = (k < lo || k >= hi) ? 1e-14f : row[k];
      row[k] = val;
      mx = fmaxf(mx, val);
    }
    float sum = 0.f;
#pragma unroll 4
    for (int k = 0; k < 129; ++k) {
      float e = expf(row[k] - mx);
      row[k] = e;
      sum += e;
    }
    const float inv = 1.f / sum;
    uint32_t* arow = (uint32_t*)(a_sh + (size_t)r * 336);
#pragma unroll 4
    for (int k2 = 0; k2 < KP / 2; ++k2) {
      float a0 = 0.f, a1 = 0.f;
      int k = 2 * k2;
      if (k <= 128) { float d = (float)(s + k) - pv; a0 = row[k] * inv * expf(-d * d * (1.f / 2048.f)); }
      ++k;
      if (k <= 128) { float d = (float)(s + k) - pv; a1 = row[k] * inv * expf(-d * d * (1.f / 2048.f)); }
      arow[k2] = pack2bf(a0, a1);
    }
  }
  __syncthreads();

  // -------- Phase B: c[64][H] = a[64][160] . selection[160][H] -------------
  for (int h0 = 0; h0 < H; h0 += 64) {
    __syncthreads();
    for (int idx = tid; idx < (KP / 2) * 64; idx += 288) {  // selection cols as B matrix
      int k2 = idx >> 6, n = idx & 63;
      int w0 = 2 * k2, w1i = 2 * k2 + 1;
      float f0 = (w0  <= 128) ? enc[((size_t)(s0 + w0)  * Bsz + b) * H + h0 + n] : 0.f;
      float f1 = (w1i <= 128) ? enc[((size_t)(s0 + w1i) * Bsz + b) * H + h0 + n] : 0.f;
      // tile = (kk-chunk, n-tile); fragment-native placement inside tile.
      ((uint32_t*)BpB)[((k2 >> 4) * 4 + (n >> 4)) * 256 + bfrag_word(k2 & 15, n & 15)] =
          pack2bf(f0, f1);
    }
    __syncthreads();
    if (w < 8) {
      const int mt = w >> 1, nt = (w & 1) * 2;   // wave covers tiles (mt,nt),(mt,nt+1)
      FragC c0, c1;
#pragma unroll
      for (int v = 0; v < 8; ++v) { c0.f[v] = 0.f; c1.f[v] = 0.f; }
      // Ping-pong pipelined over the 5 K32 chunks.
      FragAB af2[2], bA[2], bB[2];
      load_a_frag(af2[0], a_sh, mt * 16, 336, 0, lane);
      load_b_frag_n(bA[0], BpB + (0 * 4 + nt) * 1024, lane);
      load_b_frag_n(bB[0], BpB + (0 * 4 + nt + 1) * 1024, lane);
#pragma unroll
      for (int kk = 0; kk < KP / 32; ++kk) {
        const int pn = (kk + 1) & 1, pc = kk & 1;
        if (kk + 1 < KP / 32) {
          load_a_frag(af2[pn], a_sh, mt * 16, 336, (kk + 1) * 64, lane);
          load_b_frag_n(bA[pn], BpB + ((kk + 1) * 4 + nt) * 1024, lane);
          load_b_frag_n(bB[pn], BpB + ((kk + 1) * 4 + nt + 1) * 1024, lane);
        }
        c0.v = wmma_bf16(af2[pc].v, bA[pc].v, c0.v);
        c1.v = wmma_bf16(af2[pc].v, bB[pc].v, c1.v);
      }
#pragma unroll
      for (int v = 0; v < 8; ++v) {
        int t = t0 + mt * 16 + v + 8 * g;
        size_t base = ((size_t)t * Bsz + b) * H + h0;
        out[base + nt * 16 + l16]       = c0.f[v];
        out[base + (nt + 1) * 16 + l16] = c1.f[v];
      }
    }
  }
}

// ---------------------------------------------------------------------------
extern "C" void kernel_launch(void* const* d_in, const int* in_sizes, int n_in,
                              void* d_out, int out_size, void* d_ws, size_t ws_size,
                              hipStream_t stream) {
  (void)in_sizes; (void)n_in; (void)out_size; (void)ws_size;
  const float* enc = (const float*)d_in[0];   // (SPAD, B, H)
  const float* dec = (const float*)d_in[1];   // (T, B, H)
  const float* w1  = (const float*)d_in[2];   // (H, H/2)
  const float* b1  = (const float*)d_in[3];   // (H/2,)
  const float* w2  = (const float*)d_in[4];   // (H/2, 1)
  const float* b2  = (const float*)d_in[5];   // (1,)
  const int* lengths = (const int*)d_in[6];   // (B,)
  float* p_ws     = (float*)d_ws;                                  // 16384 f32
  int*   start_ws = (int*)((char*)d_ws + 16384 * sizeof(float));   // 16384 i32
  float* out = (float*)d_out;                 // (T, B, H)

  mono_mlp_kernel<<<256, 256, 0, stream>>>(dec, w1, b1, w2, b2, lengths,
                                           p_ws, start_ws);
  mono_attn_kernel<<<dim3(512 / 64, 32), 288, 0, stream>>>(enc, dec, p_ws,
                                                           start_ws, lengths, out);
}
